// AdaptiveGroupCorrelationLayer_26139170963981
// MI455X (gfx1250) — matrix operand order repaired and checked by reading, not compiled, not run on hardware
//
#include <hip/hip_runtime.h>

typedef _Float16 v16h  __attribute__((ext_vector_type(16)));
typedef _Float16 half8 __attribute__((ext_vector_type(8)));
typedef float    v8f   __attribute__((ext_vector_type(8)));

#define B_      4
#define C_      256
#define H_      96
#define W_      128
#define GROUPS_ 4
#define CG_     64
#define KW_     9
#define LSTR    72   // padded half stride (64 ch + 8 pad), 144 B rows -> 16B aligned

__global__ __launch_bounds__(256, 2)
void agcl_corr_kernel(const float* __restrict__ left,
                      const float* __restrict__ right,
                      const float* __restrict__ flow,
                      float* __restrict__ out)
{
    __shared__ _Float16 Lt[W_ * LSTR];      // left row, [w][c] f16
    __shared__ _Float16 Rt[W_ * LSTR];      // warped right row, [w][c] f16
    __shared__ int   px0[W_];
    __shared__ int   py0[W_];
    __shared__ float pwx[W_];
    __shared__ float pwy[W_];
    __shared__ float Gbuf[8][2][16][16];    // per-wave D-tile staging

    const int tid  = threadIdx.x;
    const int lane = tid & 31;
    const int wave = tid >> 5;

    int blk = blockIdx.x;
    const int g = blk % GROUPS_;  blk /= GROUPS_;
    const int h = blk % H_;
    const int b = blk / H_;

    // ---- stage 0: bilinear warp params from flow (absolute coords, align_corners) ----
    if (tid < W_) {
        const int w = tid;
        float fx = flow[((b * 2 + 0) * H_ + h) * W_ + w];
        float fy = flow[((b * 2 + 1) * H_ + h) * W_ + w];
        float x = (float)w + fx;
        float y = (float)h + fy;
        float x0f = floorf(x), y0f = floorf(y);
        px0[w] = (int)x0f;
        py0[w] = (int)y0f;
        pwx[w] = x - x0f;
        pwy[w] = y - y0f;
    }
    __syncthreads();

    // ---- stage 1: stream left + gather/blend warped right into f16 LDS ([w][c]) ----
    const long baseL = ((long)(b * C_ + g * CG_) * H_ + h) * W_;
    const long baseR = (long)(b * C_ + g * CG_) * H_ * W_;
    for (int i = tid; i < CG_ * W_; i += 256) {
        const int cc = i >> 7;          // channel within group
        const int w  = i & (W_ - 1);

        float lv = left[baseL + (long)cc * (H_ * W_) + w];
        Lt[w * LSTR + cc] = (_Float16)lv;

        const int   x0 = px0[w], y0 = py0[w];
        const float wx = pwx[w], wy = pwy[w];
        const float* rp = right + baseR + (long)cc * (H_ * W_);
        const bool xv0 = (x0     >= 0) && (x0     <= W_ - 1);
        const bool xv1 = (x0 + 1 >= 0) && (x0 + 1 <= W_ - 1);
        const bool yv0 = (y0     >= 0) && (y0     <= H_ - 1);
        const bool yv1 = (y0 + 1 >= 0) && (y0 + 1 <= H_ - 1);
        const int xc0 = min(max(x0,     0), W_ - 1);
        const int xc1 = min(max(x0 + 1, 0), W_ - 1);
        const int yc0 = min(max(y0,     0), H_ - 1);
        const int yc1 = min(max(y0 + 1, 0), H_ - 1);
        float v00 = 0.f, v01 = 0.f, v10 = 0.f, v11 = 0.f;
        if (yv0) {
            const float* r0 = rp + yc0 * W_;
            if (xv0) v00 = r0[xc0];
            if (xv1) v01 = r0[xc1];
        }
        if (yv1) {
            const float* r1 = rp + yc1 * W_;
            if (xv0) v10 = r1[xc0];
            if (xv1) v11 = r1[xc1];
        }
        float top = v00 * (1.f - wx) + v01 * wx;
        float bot = v10 * (1.f - wx) + v11 * wx;
        Rt[w * LSTR + cc] = (_Float16)(top * (1.f - wy) + bot * wy);
    }
    __syncthreads();

    // ---- stage 2: banded Gram G[w,w'] = sum_c L[c,w]*Rw[c,w'] via WMMA ----
    // wave 't' owns pixel rows [16t, 16t+16); col tiles start at 16t-8 and 16t+8
    const int t    = wave;
    const int mrow = lane & 15;
    const int hi   = lane >> 4;
    const int wA   = t * 16 + mrow;
    const int offa = hi * 8;    // A: K-halves {0-7,16-23} vs {8-15,24-31}
    const int offb = hi * 8;    // B: K-halves {0-7,16-23} vs {8-15,24-31}

    v8f acc0 = {0.f, 0.f, 0.f, 0.f, 0.f, 0.f, 0.f, 0.f};
    v8f acc1 = acc0;

    #pragma unroll
    for (int cb = 0; cb < CG_; cb += 32) {
        v16h a;
        {
            half8 lo  = *reinterpret_cast<const half8*>(&Lt[wA * LSTR + cb + offa]);
            half8 hiq = *reinterpret_cast<const half8*>(&Lt[wA * LSTR + cb + offa + 16]);
            #pragma unroll
            for (int j = 0; j < 8; ++j) { a[j] = lo[j]; a[j + 8] = hiq[j]; }
        }
        #pragma unroll
        for (int ct = 0; ct < 2; ++ct) {
            int col = t * 16 - 8 + ct * 16 + mrow;
            col = min(max(col, 0), W_ - 1);         // replicate padding
            v16h bf;
            half8 b0 = *reinterpret_cast<const half8*>(&Rt[col * LSTR + cb + offb]);
            half8 b1 = *reinterpret_cast<const half8*>(&Rt[col * LSTR + cb + offb + 16]);
            #pragma unroll
            for (int j = 0; j < 8; ++j) { bf[j] = b0[j]; bf[j + 8] = b1[j]; }
            if (ct == 0)
                acc0 = __builtin_amdgcn_wmma_f32_16x16x32_f16(
                           false, a, false, bf, (short)0, acc0, false, false);
            else
                acc1 = __builtin_amdgcn_wmma_f32_16x16x32_f16(
                           false, a, false, bf, (short)0, acc1, false, false);
        }
    }

    // D layout: lane holds N=lane&15, VGPR r holds M = r + 8*(lane>>4)
    #pragma unroll
    for (int r = 0; r < 8; ++r) {
        const int m = r + hi * 8;
        Gbuf[wave][0][m][mrow] = acc0[r];
        Gbuf[wave][1][m][mrow] = acc1[r];
    }
    // Gbuf[wave] is wave-private; in-wave LDS ordering handles the RAW.

    // ---- stage 3: extract 9-wide band, mean over Cg, write out ----
    const float scale = 1.0f / (float)CG_;
    #pragma unroll
    for (int it = 0; it < 5; ++it) {
        const int idx = it * 32 + lane;   // 0..159 covers 9k x 16m
        const int k = idx >> 4;
        const int m = idx & 15;
        if (k < KW_) {
            const int j = m + k + 4;      // col offset within [16t-8, 16t+24)
            const float val = Gbuf[wave][j >> 4][m][j & 15] * scale;
            out[(((long)b * (GROUPS_ * KW_) + g * KW_ + k) * H_ + h) * W_ + t * 16 + m] = val;
        }
    }
}

extern "C" void kernel_launch(void* const* d_in, const int* in_sizes, int n_in,
                              void* d_out, int out_size, void* d_ws, size_t ws_size,
                              hipStream_t stream) {
    (void)in_sizes; (void)n_in; (void)out_size; (void)d_ws; (void)ws_size;
    const float* left  = (const float*)d_in[0];
    const float* right = (const float*)d_in[1];
    const float* flow  = (const float*)d_in[2];
    float* out = (float*)d_out;

    const int nblocks = B_ * H_ * GROUPS_;   // 1536
    agcl_corr_kernel<<<nblocks, 256, 0, stream>>>(left, right, flow, out);
}
